// STDPAttentionFixed_88940182765805
// MI455X (gfx1250) — matrix operand order repaired and checked by reading, not compiled, not run on hardware
//
#include <hip/hip_runtime.h>
#include <hip/hip_bf16.h>

typedef __attribute__((ext_vector_type(16))) _Float16 v16h;
typedef __attribute__((ext_vector_type(8)))  _Float16 v8h;
typedef __attribute__((ext_vector_type(8)))  float    v8f;

#define TAU_INV 0.05f      // 1/tau_plus
#define A_SCALE 0.00125f   // A_plus * (1/sqrt(D)) = 0.01 * 0.125
#define FAR_CUT 512        // delta beyond which kern*score == 0 numerically -> P == 1

static __device__ __forceinline__ v8f wmma_f16(v16h a, v16h b, v8f c) {
  // D = A(16x32 f16) * B(32x16 f16) + C(16x16 f32)
  return __builtin_amdgcn_wmma_f32_16x16x32_f16(false, a, false, b, (short)0, c,
                                                false, false);
}

static __device__ __forceinline__ v16h cat8(v8h lo, v8h hi) {
  v16h a;
#pragma unroll
  for (int i = 0; i < 8; ++i) { a[i] = lo[i]; a[i + 8] = hi[i]; }
  return a;
}

// Downcast Q,K to f16 (same layout) and V to f16 transposed [bh][d][t].
__global__ void __launch_bounds__(256) stdp_prep(
    const float* __restrict__ Q, const float* __restrict__ K,
    const float* __restrict__ V, _Float16* __restrict__ Qh,
    _Float16* __restrict__ Kh, _Float16* __restrict__ Vt, int T, long long n) {
  long long i = (long long)blockIdx.x * blockDim.x + threadIdx.x;
  if (i >= n) return;
  Qh[i] = (_Float16)Q[i];
  Kh[i] = (_Float16)K[i];
  int d = (int)(i & 63);        // D = 64
  long long td = i >> 6;        // bh*T + t
  int t = (int)(td % T);
  long long bh = td / T;
  Vt[(bh * 64 + d) * (long long)T + t] = (_Float16)V[i];
}

// One wave per 16-row Q tile; 32 keys per iteration; flash-style single pass.
// Key space is split into three scalar-bounded regions:
//   [0, past_end)        : all delta<=0   -> P == 1  (PV WMMA only)
//   [past_end, far_beg)  : mixed band     -> QK WMMA + exp + PV WMMA
//   [far_beg, T)         : kern underflow -> P == 1  (PV WMMA only)
__global__ void __launch_bounds__(128) stdp_attn(
    const _Float16* __restrict__ Qh, const _Float16* __restrict__ Kh,
    const _Float16* __restrict__ Vt, float* __restrict__ O, int T) {
  const int lane = threadIdx.x & 31;
  const int wave = threadIdx.x >> 5;
  const int g = lane >> 4;    // 16-lane half group
  const int r = lane & 15;
  const int bh = blockIdx.y;
  // readfirstlane: provably wave-uniform -> scalar loop bounds, s_cbranch only,
  // EXEC stays all-ones around every WMMA.
  const int qbase =
      __builtin_amdgcn_readfirstlane((blockIdx.x * 4 + wave) * 16);

  const _Float16* Qp = Qh + (size_t)bh * T * 64;
  const _Float16* Kp = Kh + (size_t)bh * T * 64;
  const _Float16* Vp = Vt + (size_t)bh * 64 * T;

  // per-wave private P-transpose buffer: 16 rows x 40 halves (padded stride)
  __shared__ __align__(16) _Float16 pbuf_all[4][16 * 40];
  _Float16* pbuf = &pbuf_all[wave][0];

  // Q A-operands (16x32 f16 A operand in this code's contiguous k order: lane holds K = {16g..16g+15})
  const _Float16* qrow = Qp + (size_t)(qbase + r) * 64;
  v16h aq0 = cat8(*(const v8h*)(qrow + 0 + 16 * g),
                  *(const v8h*)(qrow + 8 + 16 * g));
  v16h aq1 = cat8(*(const v8h*)(qrow + 32 + 16 * g),
                  *(const v8h*)(qrow + 40 + 16 * g));

  v16h pa_ones;
#pragma unroll
  for (int i = 0; i < 16; ++i) pa_ones[i] = (_Float16)1.0f;

  v8f acc0 = {}, acc1 = {}, acc2 = {}, acc3 = {};

  // Scalar region boundaries (multiples of 32).
  int past_end = (qbase >= 31) ? (((qbase - 31) >> 5) + 1) << 5 : 0;
  int far_beg = ((qbase + 15 + FAR_CUT + 31) >> 5) << 5;
  if (far_beg > T) far_beg = T;

  const _Float16* vr0 = Vp + (size_t)(r)*T + 16 * g;
  const _Float16* vr1 = Vp + (size_t)(r + 16) * T + 16 * g;
  const _Float16* vr2 = Vp + (size_t)(r + 32) * T + 16 * g;
  const _Float16* vr3 = Vp + (size_t)(r + 48) * T + 16 * g;

  auto pv_tile = [&](int kb, v16h pa) {
    v16h bv0 = *(const v16h*)(vr0 + kb);
    v16h bv1 = *(const v16h*)(vr1 + kb);
    v16h bv2 = *(const v16h*)(vr2 + kb);
    v16h bv3 = *(const v16h*)(vr3 + kb);
    acc0 = wmma_f16(pa, bv0, acc0);
    acc1 = wmma_f16(pa, bv1, acc1);
    acc2 = wmma_f16(pa, bv2, acc2);
    acc3 = wmma_f16(pa, bv3, acc3);
  };

  // Region 1: all-past, P == 1.
  for (int kb = 0; kb < past_end; kb += 32) pv_tile(kb, pa_ones);

  // Sum-of-exp contribution of ALL fast tiles, folded into one scalar add:
  // each fast 32-key tile contributes exp(0)=1 for 2 keys per lane per row.
  float dsum[8];
  const float fast_add = 2.0f * (float)((past_end + (T - far_beg)) >> 5);
#pragma unroll
  for (int v = 0; v < 8; ++v) dsum[v] = fast_add;

  // Region 2: mixed band -> full QK^T, STDP modulation, exp, PV.
  for (int kb = past_end; kb < far_beg; kb += 32) {
    const _Float16* krow0 = Kp + (size_t)(kb + r) * 64;
    const _Float16* krow1 = Kp + (size_t)(kb + 16 + r) * 64;
    v16h bk00 = *(const v16h*)(krow0 + 0 + 16 * g);
    v16h bk01 = *(const v16h*)(krow0 + 32 + 16 * g);
    v16h bk10 = *(const v16h*)(krow1 + 0 + 16 * g);
    v16h bk11 = *(const v16h*)(krow1 + 32 + 16 * g);
    __builtin_prefetch(krow0 + 128, 0, 1);  // K row kb+r+2 -> global_prefetch_b8
    v8f s0 = {}, s1 = {};
    s0 = wmma_f16(aq0, bk00, s0);
    s0 = wmma_f16(aq1, bk01, s0);
    s1 = wmma_f16(aq0, bk10, s1);
    s1 = wmma_f16(aq1, bk11, s1);
    // C layout: element = [m = v + 8g][n = r]; apply STDP kernel + exp.
#pragma unroll
    for (int v = 0; v < 8; ++v) {
      int irow = qbase + v + 8 * g;
      int d0 = (kb + r) - irow;
      int d1 = (kb + 16 + r) - irow;
      float f0 = d0 > 0 ? A_SCALE * __expf(-TAU_INV * (float)d0) : 0.0f;
      float f1 = d1 > 0 ? A_SCALE * __expf(-TAU_INV * (float)d1) : 0.0f;
      float p0 = __expf(s0[v] * f0);
      float p1 = __expf(s1[v] * f1);
      dsum[v] += p0 + p1;
      pbuf[(v + 8 * g) * 40 + r] = (_Float16)p0;       // ds_store_b16
      pbuf[(v + 8 * g) * 40 + 16 + r] = (_Float16)p1;
    }
    // LDS ops within one wave execute in order; just stop compiler reordering.
    asm volatile("" ::: "memory");
    v8h plo = *(const v8h*)(pbuf + r * 40 + 16 * g);       // ds_load_b128
    v8h phi = *(const v8h*)(pbuf + r * 40 + 8 + 16 * g);
    v16h pa = cat8(plo, phi);
    asm volatile("" ::: "memory");
    pv_tile(kb, pa);
  }

  // Region 3: kern-underflow future, P == 1.
  for (int kb = far_beg; kb < T; kb += 32) pv_tile(kb, pa_ones);

  // Row sums: reduce across the 16 lanes of each half-group (xor masks < 16).
#pragma unroll
  for (int v = 0; v < 8; ++v) {
    float s = dsum[v];
    s += __shfl_xor(s, 1);
    s += __shfl_xor(s, 2);
    s += __shfl_xor(s, 4);
    s += __shfl_xor(s, 8);
    dsum[v] = 1.0f / s;
  }

  float* op = O + ((size_t)bh * T + qbase) * 64;
#pragma unroll
  for (int v = 0; v < 8; ++v) {
    float inv = dsum[v];
    op[(size_t)(v + 8 * g) * 64 + r]      = acc0[v] * inv;
    op[(size_t)(v + 8 * g) * 64 + 16 + r] = acc1[v] * inv;
    op[(size_t)(v + 8 * g) * 64 + 32 + r] = acc2[v] * inv;
    op[(size_t)(v + 8 * g) * 64 + 48 + r] = acc3[v] * inv;
  }
}

extern "C" void kernel_launch(void* const* d_in, const int* in_sizes, int n_in,
                              void* d_out, int out_size, void* d_ws,
                              size_t ws_size, hipStream_t stream) {
  const float* Q = (const float*)d_in[0];
  const float* K = (const float*)d_in[1];
  const float* V = (const float*)d_in[2];
  const int BH = 16, D = 64;                // B=2, H=8, D=64 per reference
  const int T = in_sizes[0] / (BH * D);     // 4096
  const long long n = (long long)BH * T * D;

  _Float16* Qhp = (_Float16*)d_ws;          // 8 MB
  _Float16* Khp = Qhp + n;                  // 8 MB
  _Float16* Vtp = Khp + n;                  // 8 MB (transposed)

  const int threads = 256;
  unsigned blocks = (unsigned)((n + threads - 1) / threads);
  stdp_prep<<<dim3(blocks), threads, 0, stream>>>(Q, K, V, Qhp, Khp, Vtp, T, n);

  dim3 grid(T / 64, BH);                    // 4 waves/block, 1 Q-tile/wave
  stdp_attn<<<grid, 128, 0, stream>>>(Qhp, Khp, Vtp, (float*)d_out, T);
}
